// Model_16844861735336
// MI455X (gfx1250) — matrix operand order repaired and checked by reading, not compiled, not run on hardware
//
#include <hip/hip_runtime.h>
#include <hip/hip_bf16.h>

typedef __attribute__((ext_vector_type(16))) _Float16 v16h;
typedef __attribute__((ext_vector_type(8)))  float    v8f;

#define BB 2
#define HH 96
#define WW 96
#define NV 32
#define QPB 128                              // queries per block (8 waves x 16)
#define BLOCKS_PER_BATCH ((HH*WW)/QPB)       // 72

// workspace float offsets
#define WS_POS 0        // pos_pred [B][N][2] = 128 floats
#define WS_TV  128      // T_vel    [B][N][2] = 128 floats
#define WS_SZ  256      // sizes [N]
#define WS_WV  288      // w     [N]
#define WS_WB_BYTEOFF 1280   // B-layout f16 weights: 18 tiles x 32 lanes x 16 halves

static __device__ __forceinline__ int clampi(int v, int lo, int hi) {
    return v < lo ? lo : (v > hi ? hi : v);
}

// ---------------------------------------------------------------------------
// Position net forward + exact d/dt via dual numbers (input dim == 1).
// Also computes sizes = 0.03 + sigmoid(param), w = sin(param).
// ---------------------------------------------------------------------------
__global__ __launch_bounds__(256) void pos_net_kernel(
    const int* __restrict__ index,
    const float* __restrict__ P1, const float* __restrict__ p1,
    const float* __restrict__ Ps1, const float* __restrict__ ps1,
    const float* __restrict__ P2, const float* __restrict__ p2,
    const float* __restrict__ Ps2, const float* __restrict__ ps2,
    const float* __restrict__ P3, const float* __restrict__ p3,
    const float* __restrict__ Ps3, const float* __restrict__ ps3,
    const float* __restrict__ Pf, const float* __restrict__ pf,
    const float* __restrict__ wparam, const float* __restrict__ szparam,
    float* __restrict__ ws)
{
    __shared__ float h1[64], d1[64], h2[128], d2[128], h3[256], d3[256];
    const int tid = threadIdx.x;
    if (tid < NV) {
        ws[WS_SZ + tid] = 0.03f + 1.0f / (1.0f + __expf(-szparam[tid]));
        ws[WS_WV + tid] = __sinf(wparam[tid]);
    }
    for (int b = 0; b < BB; ++b) {
        const float t = (float)index[b];
        if (tid < 64) {
            float pre = P1[tid]*t + p1[tid];
            float o = __sinf(pre) + Ps1[tid]*t + ps1[tid];
            h1[tid] = fmaxf(o, 0.f);
            d1[tid] = (o > 0.f) ? (__cosf(pre)*P1[tid] + Ps1[tid]) : 0.f;
        }
        __syncthreads();
        if (tid < 128) {
            float pre = p2[tid], dpre = 0.f, sc = ps2[tid], dsc = 0.f;
            for (int i = 0; i < 64; ++i) {
                float wv = P2[tid*64+i], sv = Ps2[tid*64+i];
                pre += wv*h1[i]; dpre += wv*d1[i];
                sc  += sv*h1[i]; dsc  += sv*d1[i];
            }
            float o = __sinf(pre) + sc;
            h2[tid] = fmaxf(o, 0.f);
            d2[tid] = (o > 0.f) ? (__cosf(pre)*dpre + dsc) : 0.f;
        }
        __syncthreads();
        {
            float pre = p3[tid], dpre = 0.f, sc = ps3[tid], dsc = 0.f;
            for (int i = 0; i < 128; ++i) {
                float wv = P3[tid*128+i], sv = Ps3[tid*128+i];
                pre += wv*h2[i]; dpre += wv*d2[i];
                sc  += sv*h2[i]; dsc  += sv*d2[i];
            }
            float o = __sinf(pre) + sc;
            h3[tid] = fmaxf(o, 0.f);
            d3[tid] = (o > 0.f) ? (__cosf(pre)*dpre + dsc) : 0.f;
        }
        __syncthreads();
        if (tid < 64) {
            float o = pf[tid], dv = 0.f;
            for (int i = 0; i < 256; ++i) {
                float wv = Pf[tid*256+i];
                o += wv*h3[i]; dv += wv*d3[i];
            }
            ws[WS_POS + b*64 + tid] = o;     // pos_pred flat [N*2]
            ws[WS_TV  + b*64 + tid] = dv;    // T_vel    flat [N*2]
        }
        __syncthreads();
    }
}

// ---------------------------------------------------------------------------
// Pre-swizzle W2/W3/W4 (40x40 f32, row = out-feature) into padded 64x48 f16
// WMMA B-operand layout. Tile index t = (l*2+kt)*3+nt ; per lane 16 halves.
// B[k,n] = W[n*40+k]. 16-bit B layout (wave32): VGPR v holds K = (v<4?0:16)+8g+2(v&3),+1 for
// lane group g = lane/16 (N = lane%16), offset by kt*32.
// ---------------------------------------------------------------------------
__global__ __launch_bounds__(256) void wprep_kernel(
    const float* __restrict__ W2, const float* __restrict__ W3, const float* __restrict__ W4,
    _Float16* __restrict__ wB)
{
    const int idx = blockIdx.x * 256 + threadIdx.x;
    if (idx >= 18 * 512) return;
    const int tile = idx >> 9;
    const int rem  = idx & 511;
    const int lane = rem >> 4;
    const int e    = rem & 15;
    const int l  = tile / 6;
    const int kt = (tile % 6) / 3;
    const int nt = tile % 3;
    const int v  = e >> 1;
    const int hh = e & 1;
    const int g  = lane >> 4;
    const int K  = kt*32 + ((v < 4) ? 0 : 16) + g*8 + 2*(v & 3) + hh;
    const int Nn = nt*16 + (lane & 15);
    (void)g;
    const float* Wsrc = (l == 0) ? W2 : ((l == 1) ? W3 : W4);
    const float val = (K < 40 && Nn < 40) ? Wsrc[Nn*40 + K] : 0.f;
    wB[idx] = (_Float16)val;
}

// ---------------------------------------------------------------------------
// D_vel (query = vortex positions) + vel_loss, full f32 scalar MLP. Tiny.
// ---------------------------------------------------------------------------
__global__ __launch_bounds__(64) void dvel_kernel(
    const float* __restrict__ W1, const float* __restrict__ b1,
    const float* __restrict__ Ws1, const float* __restrict__ bs1,
    const float* __restrict__ W2, const float* __restrict__ b2,
    const float* __restrict__ W3, const float* __restrict__ b3,
    const float* __restrict__ W4, const float* __restrict__ b4,
    const float* __restrict__ Wf, const float* __restrict__ bfp,
    const float* __restrict__ ws, float* __restrict__ lossPtr)
{
    __shared__ float  sW[3][1600];
    __shared__ float4 sLP[40];
    __shared__ float  sb[3][40];
    __shared__ float  sWf[40];
    __shared__ float  th[64][41];
    __shared__ float  th2[64][41];
    __shared__ float  red[64];
    const int tid = threadIdx.x;
    for (int i = tid; i < 1600; i += 64) {
        sW[0][i] = W2[i]; sW[1][i] = W3[i]; sW[2][i] = W4[i];
    }
    if (tid < 40) {
        sLP[tid] = make_float4(W1[tid], b1[tid], Ws1[tid], bs1[tid]);
        sb[0][tid] = b2[tid]; sb[1][tid] = b3[tid]; sb[2][tid] = b4[tid];
        sWf[tid] = Wf[tid];
    }
    __syncthreads();
    const int b = tid >> 5, i0 = tid & 31;
    const float px = ws[WS_POS + b*64 + i0*2];
    const float py = ws[WS_POS + b*64 + i0*2 + 1];
    const float bfv = bfp[0];
    float vx = 0.f, vy = 0.f;
    for (int n = 0; n < NV; ++n) {
        const float dx = px - ws[WS_POS + b*64 + n*2];
        const float dy = py - ws[WS_POS + b*64 + n*2 + 1];
        const float xin = sqrtf(dx*dx + dy*dy) / (ws[WS_SZ + n] * 0.5f);
        float* h  = &th[tid][0];
        float* h2 = &th2[tid][0];
        for (int j = 0; j < 40; ++j) {
            float4 p4 = sLP[j];
            float pre = p4.x*xin + p4.y;
            h[j] = fmaxf(__sinf(pre) + p4.z*xin + p4.w, 0.f);
        }
        for (int l = 0; l < 3; ++l) {
            for (int j = 0; j < 40; ++j) {
                float s = sb[l][j];
                for (int k = 0; k < 40; ++k) s += sW[l][j*40+k] * h[k];
                h2[j] = fmaxf(__sinf(s) + h[j], 0.f);
            }
            float* tsw = h; h = h2; h2 = tsw;
        }
        float mag = bfv;
        for (int j = 0; j < 40; ++j) mag += sWf[j] * h[j];
        const float wm = ws[WS_WV + n] * mag;
        vx += wm * (-dy);
        vy += wm * dx;
    }
    const float tvx = ws[WS_TV + b*64 + i0*2];
    const float tvy = ws[WS_TV + b*64 + i0*2 + 1];
    red[tid] = (tvx - vx)*(tvx - vx) + (tvy - vy)*(tvy - vy);
    __syncthreads();
    for (int s = 32; s > 0; s >>= 1) {
        if (tid < s) red[tid] += red[tid + s];
        __syncthreads();
    }
    if (tid == 0) lossPtr[0] = 0.001f * red[0] / 128.f;
}

// ---------------------------------------------------------------------------
// Main kernel: grid velocity via WMMA-batched dynamics MLP + semi-Lagrangian
// advection with bilinear sampling. One wave handles 16 query rows; the three
// 40x40 layers are v_wmma_f32_16x16x32_f16 with K padded to 64, N to 48.
// All 18 B tiles live in VGPRs for the whole kernel.
// ---------------------------------------------------------------------------
__global__ __launch_bounds__(256) void fluid_main_kernel(
    const float* __restrict__ img,
    const float* __restrict__ W1, const float* __restrict__ b1,
    const float* __restrict__ Ws1, const float* __restrict__ bs1,
    const float* __restrict__ b2, const float* __restrict__ b3, const float* __restrict__ b4,
    const float* __restrict__ Wf, const float* __restrict__ bfp,
    const float* __restrict__ ws, const _Float16* __restrict__ wB,
    float* __restrict__ out)
{
    __shared__ float  sh_h[8][16*48];       // per-wave activation scratch
    __shared__ float4 sh_l1[40];            // {W1,b1,Ws1,bs1}
    __shared__ float  sh_b[3][48];
    __shared__ float  sh_wf[40];
    __shared__ float  sh_bf;
    __shared__ float  sh_px[NV], sh_py[NV], sh_sz[NV], sh_wv[NV];

    const int tid  = threadIdx.x;
    const int lane = tid & 31;
    const int wave = tid >> 5;
    const int b    = blockIdx.x / BLOCKS_PER_BATCH;

    if (tid < 40) sh_l1[tid] = make_float4(W1[tid], b1[tid], Ws1[tid], bs1[tid]);
    if (tid < 48) {
        sh_b[0][tid] = (tid < 40) ? b2[tid] : 0.f;
        sh_b[1][tid] = (tid < 40) ? b3[tid] : 0.f;
        sh_b[2][tid] = (tid < 40) ? b4[tid] : 0.f;
    }
    if (tid >= 64 && tid < 104) sh_wf[tid-64] = Wf[tid-64];
    if (tid == 104) sh_bf = bfp[0];
    if (tid >= 128 && tid < 160) {
        const int n = tid - 128;
        sh_px[n] = ws[WS_POS + b*64 + n*2];
        sh_py[n] = ws[WS_POS + b*64 + n*2 + 1];
        sh_sz[n] = ws[WS_SZ + n];
        sh_wv[n] = ws[WS_WV + n];
    }
    __syncthreads();

    // resident B-operand weight tiles: [layer][ktile][ntile] -> (l*2+kt)*3+nt
    v16h Bw[18];
    #pragma unroll
    for (int t = 0; t < 18; ++t)
        Bw[t] = *(const v16h*)(wB + (t*32 + lane)*16);

    const int r = lane & 15;        // matrix row owned by this lane
    const int g = lane >> 4;        // half-wave group (A-layout K split)
    const int q = (blockIdx.x % BLOCKS_PER_BATCH)*QPB + wave*16 + r;
    const int colq = q % WW, rowq = q / WW;
    const float qx = (colq + 0.5f) / (float)WW;
    const float qy = (rowq + 0.5f) / (float)HH;

    float velx = 0.f, vely = 0.f;
    float* hb = &sh_h[wave][0];

    for (int n = 0; n < NV; ++n) {
        const float dxv = qx - sh_px[n];
        const float dyv = qy - sh_py[n];
        const float xin = sqrtf(dxv*dxv + dyv*dyv) / (sh_sz[n] * 0.5f);

        // ---- layer 1 (scalar input): compute straight into A-layout regs +
        //      store f32 copy to LDS for the layer-2 shortcut.
        // A 16x32 f16 layout: VGPR v, group g: K = (v<4?0:16) + g*8 + 2*(v&3) + e
        v16h a0, a1;
        #pragma unroll
        for (int v = 0; v < 8; ++v) {
            const int kb = ((v < 4) ? 0 : 16) + g*8 + 2*(v & 3);
            #pragma unroll
            for (int e = 0; e < 2; ++e) {
                const int j = kb + e;
                const float4 p4 = sh_l1[j];
                const float pre = p4.x * xin + p4.y;
                const float hv = fmaxf(__sinf(pre) + p4.z * xin + p4.w, 0.f);
                a0[2*v+e] = (_Float16)hv;
                hb[r*48 + j] = hv;
                const int j1 = j + 32;
                float hv1 = 0.f;
                if (j1 < 40) {
                    const float4 q4 = sh_l1[j1];
                    const float pre1 = q4.x * xin + q4.y;
                    hv1 = fmaxf(__sinf(pre1) + q4.z * xin + q4.w, 0.f);
                    hb[r*48 + j1] = hv1;
                }
                a1[2*v+e] = (_Float16)hv1;
            }
        }
        __builtin_amdgcn_wave_barrier();

        // ---- layers 2..4: WMMA GEMM + sin + shortcut + relu
        #pragma unroll
        for (int l = 0; l < 3; ++l) {
            v8f c[3];
            #pragma unroll
            for (int nt = 0; nt < 3; ++nt) {
                v8f acc = {0.f,0.f,0.f,0.f,0.f,0.f,0.f,0.f};
                acc = __builtin_amdgcn_wmma_f32_16x16x32_f16(
                        false, a0, false, Bw[(l*2+0)*3+nt], (short)0, acc, false, false);
                acc = __builtin_amdgcn_wmma_f32_16x16x32_f16(
                        false, a1, false, Bw[(l*2+1)*3+nt], (short)0, acc, false, false);
                c[nt] = acc;
            }
            // writeback in C layout: lane holds col (nt*16 + r), rows g*8..g*8+7
            const int rbase = g*8;
            #pragma unroll
            for (int nt = 0; nt < 3; ++nt) {
                const int colj = nt*16 + r;
                #pragma unroll
                for (int k = 0; k < 8; ++k) {
                    const int row = rbase + k;
                    float nv = 0.f;
                    if (colj < 40) {
                        const float pre = c[nt][k] + sh_b[l][colj];
                        nv = fmaxf(__sinf(pre) + hb[row*48 + colj], 0.f);
                    }
                    hb[row*48 + colj] = nv;
                }
            }
            __builtin_amdgcn_wave_barrier();
            if (l < 2) {
                // rebuild A-layout operands from LDS (wave-private; DS in-order)
                #pragma unroll
                for (int v = 0; v < 8; ++v) {
                    const int kb = ((v < 4) ? 0 : 16) + g*8 + 2*(v & 3);
                    a0[2*v]   = (_Float16)hb[r*48 + kb];
                    a0[2*v+1] = (_Float16)hb[r*48 + kb + 1];
                    const int k1 = kb + 32;
                    const float t0 = (k1   < 40) ? hb[r*48 + k1]     : 0.f;
                    const float t1 = (k1+1 < 40) ? hb[r*48 + k1 + 1] : 0.f;
                    a1[2*v]   = (_Float16)t0;
                    a1[2*v+1] = (_Float16)t1;
                }
                __builtin_amdgcn_wave_barrier();
            }
        }

        // ---- final 40 -> 1 dot
        float mag = sh_bf;
        #pragma unroll
        for (int j = 0; j < 40; ++j)
            mag += hb[r*48 + j] * sh_wf[j];

        const float wm = sh_wv[n] * mag;
        velx += wm * (-dyv);
        vely += wm * dxv;
    }

    // ---- semi-Lagrangian backtrace + bilinear sample (lanes 0..15 own rows)
    if (lane < 16) {
        const float bx = qx - 0.01f * velx;
        const float by = qy - 0.01f * vely;
        const float pxf = bx * (float)WW - 0.5f;
        const float pyf = by * (float)HH - 0.5f;
        const float fx = floorf(pxf), fy = floorf(pyf);
        const float wx = pxf - fx, wy = pyf - fy;
        const int x0 = clampi((int)fx, 0, WW-1);
        const int x1 = (x0 + 1 < WW) ? x0 + 1 : WW-1;
        const int y0 = clampi((int)fy, 0, HH-1);
        const int y1 = (y0 + 1 < HH) ? y0 + 1 : HH-1;
        const float* ib = img + b*HH*WW*3;
        #pragma unroll
        for (int cch = 0; cch < 3; ++cch) {
            const float v00 = ib[(y0*WW+x0)*3+cch];
            const float v01 = ib[(y0*WW+x1)*3+cch];
            const float v10 = ib[(y1*WW+x0)*3+cch];
            const float v11 = ib[(y1*WW+x1)*3+cch];
            const float top = v00*(1.f-wx) + v01*wx;
            const float bot = v10*(1.f-wx) + v11*wx;
            out[((b*HH + rowq)*WW + colq)*3 + cch] = top*(1.f-wy) + bot*wy;
        }
    }
}

// ---------------------------------------------------------------------------
extern "C" void kernel_launch(void* const* d_in, const int* in_sizes, int n_in,
                              void* d_out, int out_size, void* d_ws, size_t ws_size,
                              hipStream_t stream)
{
    (void)in_sizes; (void)n_in; (void)ws_size;
    const float* ximg = (const float*)d_in[0];
    const int*   idx  = (const int*)d_in[1];
    const float* W1  = (const float*)d_in[2];
    const float* b1  = (const float*)d_in[3];
    const float* Ws1 = (const float*)d_in[4];
    const float* bs1 = (const float*)d_in[5];
    const float* W2  = (const float*)d_in[6];
    const float* b2  = (const float*)d_in[7];
    const float* W3  = (const float*)d_in[8];
    const float* b3  = (const float*)d_in[9];
    const float* W4  = (const float*)d_in[10];
    const float* b4  = (const float*)d_in[11];
    const float* Wf  = (const float*)d_in[12];
    const float* bf  = (const float*)d_in[13];
    const float* P1  = (const float*)d_in[14];
    const float* p1  = (const float*)d_in[15];
    const float* Ps1 = (const float*)d_in[16];
    const float* ps1 = (const float*)d_in[17];
    const float* P2  = (const float*)d_in[18];
    const float* p2  = (const float*)d_in[19];
    const float* Ps2 = (const float*)d_in[20];
    const float* ps2 = (const float*)d_in[21];
    const float* P3  = (const float*)d_in[22];
    const float* p3  = (const float*)d_in[23];
    const float* Ps3 = (const float*)d_in[24];
    const float* ps3 = (const float*)d_in[25];
    const float* Pf  = (const float*)d_in[26];
    const float* pf  = (const float*)d_in[27];
    const float* wp  = (const float*)d_in[28];
    const float* sp  = (const float*)d_in[29];

    float*    wsf = (float*)d_ws;
    _Float16* wBt = (_Float16*)((char*)d_ws + WS_WB_BYTEOFF);
    float*    out = (float*)d_out;
    float*    lossPtr = out + (out_size - 1);

    pos_net_kernel<<<1, 256, 0, stream>>>(idx, P1,p1,Ps1,ps1, P2,p2,Ps2,ps2,
                                          P3,p3,Ps3,ps3, Pf,pf, wp, sp, wsf);
    wprep_kernel<<<36, 256, 0, stream>>>(W2, W3, W4, wBt);
    dvel_kernel<<<1, 64, 0, stream>>>(W1,b1,Ws1,bs1, W2,b2, W3,b3, W4,b4,
                                      Wf,bf, wsf, lossPtr);
    fluid_main_kernel<<<BB*BLOCKS_PER_BATCH, 256, 0, stream>>>(
        ximg, W1,b1,Ws1,bs1, b2,b3,b4, Wf, bf, wsf, wBt, out);
}
